// CMPNEncoder_8607114461814
// MI455X (gfx1250) — matrix operand order repaired and checked by reading, not compiled, not run on hardware
//
#include <hip/hip_runtime.h>
#include <math.h>

// ---------------- problem constants (from reference) ----------------
#define H   300
#define H2  600
#define H3  900
#define NA  51201      // n_atoms
#define NAP 51216      // padded to 16
#define NB  102401     // n_bonds
#define NBP 102416     // padded to 16
#define BMOL 1024      // molecules
#define AMOL 50        // atoms per molecule
#define NBA 51200      // BMOL*AMOL

typedef __bf16 bf_t;
typedef __attribute__((ext_vector_type(16))) __bf16 v16bf;
typedef __attribute__((ext_vector_type(8)))  __bf16 v8bf;
typedef __attribute__((ext_vector_type(8)))  float  v8f;

// ---------------- WMMA bf16 GEMM ----------------
// C[M x N] = epilogue( A[Mpad x Kpad](bf16) @ Bt[Npad x Kpad]^T (bf16)
//                      + bias[n] + addsrc[m*N+n] ), epilogue = optional ReLU.
// Each wave computes a 16(M) x 64(N) tile: one A fragment reused on 4 B tiles.
// Fragment layouts per CDNA5 ISA 7.12.2 (wave32):
//   A 16x32: lane l: row M = l&15; halves 0..7 = K + 8*(l>>4) .. +7 ;
//                                  halves 8..15 = K+16+8*(l>>4) .. +7
//   B 32x16: lane l: col N = l&15; halves 0..7 = K + 8*(l>>4) .. +7 ;
//                                  halves 8..15 = K+16+8*(l>>4) .. +7
//   C/D f32 16x16: lane l: col N = l&15; vgpr v: row M = v + 8*(l>>4)

__device__ __forceinline__ v16bf load_frag_a(const bf_t* __restrict__ arow,
                                             int k, int lhalf) {
  v8bf lo = *(const v8bf*)(arow + k + 8 * lhalf);
  v8bf hi = *(const v8bf*)(arow + k + 16 + 8 * lhalf);
  v16bf r;
#pragma unroll
  for (int i = 0; i < 8; ++i) { r[i] = lo[i]; r[i + 8] = hi[i]; }
  return r;
}

__device__ __forceinline__ v16bf load_frag_b(const bf_t* __restrict__ brow,
                                             int k, int lhalf) {
  const int kb = k + 8 * lhalf;
  v8bf lo = *(const v8bf*)(brow + kb);
  v8bf hi = *(const v8bf*)(brow + kb + 16);
  v16bf r;
#pragma unroll
  for (int i = 0; i < 8; ++i) { r[i] = lo[i]; r[i + 8] = hi[i]; }
  return r;
}

__device__ __forceinline__ void store_tile(float* __restrict__ C, v8f acc,
                                           int tM, int nCol, int M, int N,
                                           const float* __restrict__ bias,
                                           const float* __restrict__ addsrc,
                                           int do_relu, int lhalf) {
  if (nCol >= N) return;
  const float bv = bias ? bias[nCol] : 0.0f;
#pragma unroll
  for (int v = 0; v < 8; ++v) {
    const int m = tM + v + 8 * lhalf;
    if (m < M) {
      float val = acc[v] + bv;
      if (addsrc) val += addsrc[(size_t)m * N + nCol];
      if (do_relu) val = fmaxf(val, 0.0f);
      C[(size_t)m * N + nCol] = val;
    }
  }
}

__global__ void wmma_gemm_bf16(const bf_t* __restrict__ A,
                               const bf_t* __restrict__ Bt,
                               float* __restrict__ C,
                               const float* __restrict__ bias,
                               const float* __restrict__ addsrc,
                               int M, int N, int Kpad, int tilesN,
                               int totalWaves, int do_relu) {
  const int wid = (int)(blockIdx.x * (blockDim.x >> 5) + (threadIdx.x >> 5));
  if (wid >= totalWaves) return;              // wave-uniform: EXEC stays all-1s
  const int lane  = threadIdx.x & 31;
  const int lhalf = lane >> 4;
  const int l15   = lane & 15;
  const int tM = (wid / tilesN) * 16;
  const int tN = (wid % tilesN) * 64;

  const bf_t* arow  = A  + (size_t)(tM + l15) * Kpad;
  const bf_t* brow0 = Bt + (size_t)(tN + l15) * Kpad;
  const size_t bstr = (size_t)16 * Kpad;

  v8f acc0 = {}, acc1 = {}, acc2 = {}, acc3 = {};
  for (int k = 0; k < Kpad; k += 32) {
    __builtin_prefetch(arow + k + 64, 0, 1);   // global_prefetch_b8 ahead on A
    v16bf a  = load_frag_a(arow, k, lhalf);
    v16bf b0 = load_frag_b(brow0,            k, lhalf);
    v16bf b1 = load_frag_b(brow0 + bstr,     k, lhalf);
    v16bf b2 = load_frag_b(brow0 + 2 * bstr, k, lhalf);
    v16bf b3 = load_frag_b(brow0 + 3 * bstr, k, lhalf);
    acc0 = __builtin_amdgcn_wmma_f32_16x16x32_bf16(false, a, false, b0, (short)0, acc0, false, false);
    acc1 = __builtin_amdgcn_wmma_f32_16x16x32_bf16(false, a, false, b1, (short)0, acc1, false, false);
    acc2 = __builtin_amdgcn_wmma_f32_16x16x32_bf16(false, a, false, b2, (short)0, acc2, false, false);
    acc3 = __builtin_amdgcn_wmma_f32_16x16x32_bf16(false, a, false, b3, (short)0, acc3, false, false);
  }
  store_tile(C, acc0, tM, tN +  0 + l15, M, N, bias, addsrc, do_relu, lhalf);
  store_tile(C, acc1, tM, tN + 16 + l15, M, N, bias, addsrc, do_relu, lhalf);
  store_tile(C, acc2, tM, tN + 32 + l15, M, N, bias, addsrc, do_relu, lhalf);
  store_tile(C, acc3, tM, tN + 48 + l15, M, N, bias, addsrc, do_relu, lhalf);
}

// ---------------- operand packing (f32 -> zero-padded bf16 panels) ----------------
__global__ void pack_bf16_kernel(bf_t* __restrict__ dst, const float* __restrict__ src,
                                 int M, int K, long total, int Kpad) {
  const long idx = (long)blockIdx.x * blockDim.x + threadIdx.x;
  if (idx >= total) return;
  const int m = (int)(idx / Kpad), k = (int)(idx % Kpad);
  const float v = (m < M && k < K) ? src[(size_t)m * K + k] : 0.0f;
  dst[idx] = (bf_t)v;
}

// weight -> Bt[Npad x Kpad], transpose mode: Bt[n][k] = W[k*N + n]
__global__ void prep_wT(bf_t* __restrict__ dst, const float* __restrict__ src,
                        int K, int N, long total, int Kpad) {
  const long idx = (long)blockIdx.x * blockDim.x + threadIdx.x;
  if (idx >= total) return;
  const int n = (int)(idx / Kpad), k = (int)(idx % Kpad);
  const float v = (n < N && k < K) ? src[(size_t)k * N + n] : 0.0f;
  dst[idx] = (bf_t)v;
}
// weight -> Bt[Npad x Kpad], copy mode (for x@W^T): Bt[n][k] = W[n*K + k]
__global__ void prep_wC(bf_t* __restrict__ dst, const float* __restrict__ src,
                        int Nn, int K, long total, int Kpad) {
  const long idx = (long)blockIdx.x * blockDim.x + threadIdx.x;
  if (idx >= total) return;
  const int n = (int)(idx / Kpad), k = (int)(idx % Kpad);
  const float v = (n < Nn && k < K) ? src[(size_t)n * K + k] : 0.0f;
  dst[idx] = (bf_t)v;
}

// ---------------- message-passing elementwise kernels ----------------
__global__ void gather_sum_max(float* __restrict__ out, const float* __restrict__ mbond,
                               const int* __restrict__ a2b, int accumulate) {
  const long idx = (long)blockIdx.x * blockDim.x + threadIdx.x;
  if (idx >= (long)NA * H) return;
  const int a = (int)(idx / H), h = (int)(idx % H);
  float s = 0.0f, mx = -INFINITY;
#pragma unroll
  for (int j = 0; j < 6; ++j) {
    const int b = a2b[a * 6 + j];
    const float v = mbond[(size_t)b * H + h];
    s += v; mx = fmaxf(mx, v);
  }
  const float r = s * mx;
  if (accumulate) out[idx] += r; else out[idx] = r;
}

// packed bf16 [(NBP)x320] of (message_atom[b2a[b]] - message_bond[b2revb[b]])
__global__ void bond_diff_pack(bf_t* __restrict__ dst, const float* __restrict__ matom,
                               const float* __restrict__ mbond,
                               const int* __restrict__ b2a, const int* __restrict__ b2revb) {
  const long idx = (long)blockIdx.x * blockDim.x + threadIdx.x;
  if (idx >= (long)NBP * 320) return;
  const int m = (int)(idx / 320), k = (int)(idx % 320);
  float v = 0.0f;
  if (m < NB && k < H)
    v = matom[(size_t)b2a[m] * H + k] - mbond[(size_t)b2revb[m] * H + k];
  dst[idx] = (bf_t)v;
}

// concat [agg | message_atom | input_atom] -> bf16 [NAP x 928]
__global__ void pack_concat3(bf_t* __restrict__ dst, const float* __restrict__ agg,
                             const float* __restrict__ matom, const float* __restrict__ iatom) {
  const long idx = (long)blockIdx.x * blockDim.x + threadIdx.x;
  if (idx >= (long)NAP * 928) return;
  const int m = (int)(idx / 928), k = (int)(idx % 928);
  float v = 0.0f;
  if (m < NA) {
    if (k < H)        v = agg  [(size_t)m * H + k];
    else if (k < H2)  v = matom[(size_t)m * H + (k - H)];
    else if (k < H3)  v = iatom[(size_t)m * H + (k - H2)];
  }
  dst[idx] = (bf_t)v;
}

__global__ void bias_relu(float* __restrict__ msg, const float* __restrict__ hidden,
                          const float* __restrict__ bias) {
  const long idx = (long)blockIdx.x * blockDim.x + threadIdx.x;
  if (idx >= (long)NA * H) return;
  msg[idx] = fmaxf(hidden[idx] + bias[idx % H], 0.0f);
}

__global__ void h0_max(float* __restrict__ h0, const float* __restrict__ hidden) {
  const long idx = (long)blockIdx.x * blockDim.x + threadIdx.x;
  if (idx >= (long)BMOL * H) return;
  const int b = (int)(idx / H), h = (int)(idx % H);
  float mx = -INFINITY;
  for (int a = 0; a < AMOL; ++a)
    mx = fmaxf(mx, hidden[(size_t)(1 + b * AMOL + a) * H + h]);
  h0[idx] = mx;
}

__device__ __forceinline__ float sigmoidf_(float x) { return 1.0f / (1.0f + expf(-x)); }

__global__ void gru_cell(float* __restrict__ h, const float* __restrict__ gi_all,
                         const float* __restrict__ gh, float* __restrict__ gru_out,
                         int t, int dir) {
  const long idx = (long)blockIdx.x * blockDim.x + threadIdx.x;
  if (idx >= (long)BMOL * H) return;
  const int b = (int)(idx / H), hh = (int)(idx % H);
  const float* gi = gi_all + ((size_t)b * AMOL + t) * H3;
  const float* g  = gh + (size_t)b * H3;
  const float r = sigmoidf_(gi[hh] + g[hh]);
  const float z = sigmoidf_(gi[H + hh] + g[H + hh]);
  const float n = tanhf(gi[2 * H + hh] + r * g[2 * H + hh]);
  const float hn = (1.0f - z) * n + z * h[idx];
  h[idx] = hn;
  gru_out[((size_t)b * AMOL + t) * H2 + dir * H + hh] = hn;
}

// agg_message -> bf16 [NAP x 608]; row0 = [msg0|msg0], rows i>=1 = gru_out[i-1]
__global__ void pack_aggmsg(bf_t* __restrict__ dst, const float* __restrict__ msg,
                            const float* __restrict__ gru_out) {
  const long idx = (long)blockIdx.x * blockDim.x + threadIdx.x;
  if (idx >= (long)NAP * 608) return;
  const int m = (int)(idx / 608), k = (int)(idx % 608);
  float v = 0.0f;
  if (m == 0) { if (k < H) v = msg[k]; else if (k < H2) v = msg[k - H]; }
  else if (m < NA && k < H2) v = gru_out[(size_t)(m - 1) * H2 + k];
  dst[idx] = (bf_t)v;
}

__global__ void mean_pool(float* __restrict__ out, const float* __restrict__ ah) {
  const long idx = (long)blockIdx.x * blockDim.x + threadIdx.x;
  if (idx >= (long)BMOL * H) return;
  const int b = (int)(idx / H), h = (int)(idx % H);
  float s = 0.0f;
  for (int a = 0; a < AMOL; ++a) s += ah[(size_t)(1 + b * AMOL + a) * H + h];
  out[idx] = s * (1.0f / AMOL);
}

// ---------------- host side ----------------
static inline int g1(long n) { return (int)((n + 255) / 256); }

static void launch_gemm(const bf_t* A, const bf_t* Bt, float* C, const float* bias,
                        const float* addsrc, int M, int Mpad, int N, int Npad,
                        int Kpad, int relu, hipStream_t s) {
  const int tilesN = Npad / 64;
  const int tw = (Mpad / 16) * tilesN;
  wmma_gemm_bf16<<<(tw + 7) / 8, 256, 0, s>>>(A, Bt, C, bias, addsrc,
                                              M, N, Kpad, tilesN, tw, relu);
}

extern "C" void kernel_launch(void* const* d_in, const int* in_sizes, int n_in,
                              void* d_out, int out_size, void* d_ws, size_t ws_size,
                              hipStream_t stream) {
  const float* f_atoms  = (const float*)d_in[0];
  const float* f_bonds  = (const float*)d_in[1];
  const int*   a2b      = (const int*)d_in[2];
  const int*   b2a      = (const int*)d_in[3];
  const int*   b2revb   = (const int*)d_in[4];
  const float* W_i_atom = (const float*)d_in[5];
  const float* W_i_bond = (const float*)d_in[6];
  const float* W_h      = (const float*)d_in[7];
  const float* W_lr     = (const float*)d_in[8];
  const float* gru_bias = (const float*)d_in[9];
  const float* w_ih_f   = (const float*)d_in[10];
  const float* w_hh_f   = (const float*)d_in[11];
  const float* b_ih_f   = (const float*)d_in[12];
  const float* b_hh_f   = (const float*)d_in[13];
  const float* w_ih_b   = (const float*)d_in[14];
  const float* w_hh_b   = (const float*)d_in[15];
  const float* b_ih_b   = (const float*)d_in[16];
  const float* b_hh_b   = (const float*)d_in[17];
  const float* W_o      = (const float*)d_in[18];
  const float* b_o      = (const float*)d_in[19];

  // linear scratch allocator (~1.16 GB needed)
  char* p = (char*)d_ws;
  auto alloc = [&](size_t bytes) -> char* {
    char* r = p; p += (bytes + 255) & ~(size_t)255; return r;
  };
  float* input_atom   = (float*)alloc((size_t)NA * H * 4);
  float* message_atom = (float*)alloc((size_t)NA * H * 4);
  float* input_bond   = (float*)alloc((size_t)NB * H * 4);
  float* bond0        = (float*)alloc((size_t)NB * H * 4);
  float* bond1        = (float*)alloc((size_t)NB * H * 4);
  float* aggb         = (float*)alloc((size_t)NA * H * 4);
  float* hidden       = (float*)alloc((size_t)NA * H * 4);
  float* msg          = (float*)alloc((size_t)NA * H * 4);
  float* atom_hid     = (float*)alloc((size_t)NA * H * 4);
  float* h0           = (float*)alloc((size_t)BMOL * H * 4);
  float* hstate       = (float*)alloc((size_t)BMOL * H * 4);
  float* gh           = (float*)alloc((size_t)BMOL * H3 * 4);
  float* gi_all       = (float*)alloc((size_t)NBA * H3 * 4);
  float* gru_out      = (float*)alloc((size_t)NBA * H2 * 4);
  bf_t*  packA        = (bf_t*)alloc((size_t)NAP * 928 * 2);  // largest panel
  bf_t*  packH        = (bf_t*)alloc((size_t)BMOL * 320 * 2);
  bf_t*  wt_atom = (bf_t*)alloc((size_t)320 * 160 * 2);
  bf_t*  wt_bond = (bf_t*)alloc((size_t)320 * 160 * 2);
  bf_t*  wt_Wh   = (bf_t*)alloc((size_t)3 * 320 * 320 * 2);
  bf_t*  wt_lr   = (bf_t*)alloc((size_t)320 * 928 * 2);
  bf_t*  wt_ihf  = (bf_t*)alloc((size_t)960 * 320 * 2);
  bf_t*  wt_hhf  = (bf_t*)alloc((size_t)960 * 320 * 2);
  bf_t*  wt_ihb  = (bf_t*)alloc((size_t)960 * 320 * 2);
  bf_t*  wt_hhb  = (bf_t*)alloc((size_t)960 * 320 * 2);
  bf_t*  wt_o    = (bf_t*)alloc((size_t)320 * 608 * 2);

  // --- weight prep (L2-resident bf16, K padded to 32, N padded to 64) ---
  prep_wT<<<g1(320L*160), 256, 0, stream>>>(wt_atom, W_i_atom, 133, 300, 320L*160, 160);
  prep_wT<<<g1(320L*160), 256, 0, stream>>>(wt_bond, W_i_bond, 147, 300, 320L*160, 160);
  for (int d = 0; d < 3; ++d)
    prep_wT<<<g1(320L*320), 256, 0, stream>>>(wt_Wh + (size_t)d*320*320,
                                              W_h + (size_t)d*H*H, 300, 300, 320L*320, 320);
  prep_wT<<<g1(320L*928), 256, 0, stream>>>(wt_lr, W_lr, 900, 300, 320L*928, 928);
  prep_wC<<<g1(960L*320), 256, 0, stream>>>(wt_ihf, w_ih_f, 900, 300, 960L*320, 320);
  prep_wC<<<g1(960L*320), 256, 0, stream>>>(wt_hhf, w_hh_f, 900, 300, 960L*320, 320);
  prep_wC<<<g1(960L*320), 256, 0, stream>>>(wt_ihb, w_ih_b, 900, 300, 960L*320, 320);
  prep_wC<<<g1(960L*320), 256, 0, stream>>>(wt_hhb, w_hh_b, 900, 300, 960L*320, 320);
  prep_wT<<<g1(320L*608), 256, 0, stream>>>(wt_o, W_o, 600, 300, 320L*608, 608);

  // --- input projections ---
  pack_bf16_kernel<<<g1((long)NAP*160), 256, 0, stream>>>(packA, f_atoms, NA, 133, (long)NAP*160, 160);
  launch_gemm(packA, wt_atom, input_atom, nullptr, nullptr, NA, NAP, H, 320, 160, 1, stream);
  hipMemcpyAsync(message_atom, input_atom, (size_t)NA*H*4, hipMemcpyDeviceToDevice, stream);

  pack_bf16_kernel<<<g1((long)NBP*160), 256, 0, stream>>>(packA, f_bonds, NB, 147, (long)NBP*160, 160);
  launch_gemm(packA, wt_bond, input_bond, nullptr, nullptr, NB, NBP, H, 320, 160, 1, stream);
  hipMemcpyAsync(bond0, input_bond, (size_t)NB*H*4, hipMemcpyDeviceToDevice, stream);

  // --- DEPTH-1 message passing iterations ---
  float* mb = bond0; float* mbAlt = bond1;
  for (int d = 0; d < 3; ++d) {
    gather_sum_max<<<g1((long)NA*H), 256, 0, stream>>>(message_atom, mb, a2b, 1);
    bond_diff_pack<<<g1((long)NBP*320), 256, 0, stream>>>(packA, message_atom, mb, b2a, b2revb);
    launch_gemm(packA, wt_Wh + (size_t)d*320*320, mbAlt, nullptr, input_bond,
                NB, NBP, H, 320, 320, 1, stream);
    float* t = mb; mb = mbAlt; mbAlt = t;
  }

  // --- final aggregation + W_lr ---
  gather_sum_max<<<g1((long)NA*H), 256, 0, stream>>>(aggb, mb, a2b, 0);
  pack_concat3<<<g1((long)NAP*928), 256, 0, stream>>>(packA, aggb, message_atom, input_atom);
  launch_gemm(packA, wt_lr, hidden, nullptr, nullptr, NA, NAP, H, 320, 928, 0, stream);

  // --- BatchGRU setup ---
  bias_relu<<<g1((long)NA*H), 256, 0, stream>>>(msg, hidden, gru_bias);
  h0_max<<<g1((long)BMOL*H), 256, 0, stream>>>(h0, hidden);
  // x panel (shared by both directions): msg rows 1..NBA
  pack_bf16_kernel<<<g1((long)NBA*320), 256, 0, stream>>>(packA, msg + H, NBA, H, (long)NBA*320, 320);

  // forward direction
  launch_gemm(packA, wt_ihf, gi_all, b_ih_f, nullptr, NBA, NBA, H3, 960, 320, 0, stream);
  hipMemcpyAsync(hstate, h0, (size_t)BMOL*H*4, hipMemcpyDeviceToDevice, stream);
  for (int t = 0; t < AMOL; ++t) {
    pack_bf16_kernel<<<g1((long)BMOL*320), 256, 0, stream>>>(packH, hstate, BMOL, H, (long)BMOL*320, 320);
    launch_gemm(packH, wt_hhf, gh, b_hh_f, nullptr, BMOL, BMOL, H3, 960, 320, 0, stream);
    gru_cell<<<g1((long)BMOL*H), 256, 0, stream>>>(hstate, gi_all, gh, gru_out, t, 0);
  }
  // backward direction (reverse scan; ys stay in time order)
  launch_gemm(packA, wt_ihb, gi_all, b_ih_b, nullptr, NBA, NBA, H3, 960, 320, 0, stream);
  hipMemcpyAsync(hstate, h0, (size_t)BMOL*H*4, hipMemcpyDeviceToDevice, stream);
  for (int tt = 0; tt < AMOL; ++tt) {
    const int t = AMOL - 1 - tt;
    pack_bf16_kernel<<<g1((long)BMOL*320), 256, 0, stream>>>(packH, hstate, BMOL, H, (long)BMOL*320, 320);
    launch_gemm(packH, wt_hhb, gh, b_hh_b, nullptr, BMOL, BMOL, H3, 960, 320, 0, stream);
    gru_cell<<<g1((long)BMOL*H), 256, 0, stream>>>(hstate, gi_all, gh, gru_out, t, 1);
  }

  // --- output projection + molecule mean pool ---
  pack_aggmsg<<<g1((long)NAP*608), 256, 0, stream>>>(packA, msg, gru_out);
  launch_gemm(packA, wt_o, atom_hid, b_o, nullptr, NA, NAP, H, 320, 608, 1, stream);
  mean_pool<<<g1((long)BMOL*H), 256, 0, stream>>>((float*)d_out, atom_hid);
}
